// MultiHeadAttention_1726576854477
// MI455X (gfx1250) — compile-verified, matrix operand order checked by reading, not run on hardware
//
#include <hip/hip_runtime.h>
#include <hip/hip_bf16.h>

// ============================================================================
// MI455X (gfx1250) conv3d-QKV (implicit GEMM, bf16 WMMA, TDM-staged B tiles)
// + causal attention (WMMA + wave32 shuffle softmax).
//
// Roofline: conv ~348 GFLOP is compute-bound -> all MACs via
// v_wmma_f32_16x16x32_bf16. qkv kept bf16 (96 MiB) in d_ws => L2-resident
// (192 MB L2); HBM traffic ~200 MB floor (~9us @ 23.3 TB/s). Interior conv
// input tiles are DMA'd by the Tensor Data Mover (tensor_load_to_lds,
// TENSORcnt) -- a clean 2D strided tile (32 w contiguous x 32 ci, stride
// 1024 elems); padded boundary taps fall back to VALU staging.
// ============================================================================

typedef __bf16 bf16_t;
typedef __attribute__((ext_vector_type(8)))  __bf16 v8bf;
typedef __attribute__((ext_vector_type(16))) __bf16 v16bf;
typedef __attribute__((ext_vector_type(8)))  float  v8f;
typedef unsigned int u32x4 __attribute__((ext_vector_type(4)));
typedef int i32x8 __attribute__((ext_vector_type(8)));
typedef int i32x4 __attribute__((ext_vector_type(4)));

union BF16Frag { v8bf h[2]; v16bf v; };

__device__ inline v8f vzero8() {
  v8f z = {0.f, 0.f, 0.f, 0.f, 0.f, 0.f, 0.f, 0.f};
  return z;
}

__device__ inline v8f wmma_bf16(v16bf a, v16bf b, v8f c) {
  // D = A(16x32) * B(32x16) + C(16x16 f32)
  return __builtin_amdgcn_wmma_f32_16x16x32_bf16(
      /*neg_a=*/false, a, /*neg_b=*/false, b,
      /*c_mod=*/(short)0, c, /*reuse_a=*/false, /*reuse_b=*/false);
}

// Problem constants
#define NB   4
#define NS   32
#define NC   128
#define NCO  384
#define NH   32
#define NW   32
#define NHW  1024
#define NHEAD 4
#define HD   32
#define CIK  3456
#define WS_TENSOR_ELEMS 16777216  // per q/k/v bf16 tensor (32 MiB)

// ----------------------------------------------------------------------------
// Kernel 1: conv3d implicit GEMM. Grid (3, 1024, 4), block 256 (8 waves).
// Block tile: M=128 co x N=32 w. Wave grid 4x2 (mT x nT); each wave owns two
// 16-row A fragments sharing one B fragment -> 2 WMMAs per 32-deep K step.
// ----------------------------------------------------------------------------
__global__ __launch_bounds__(256) void conv_qkv_wmma(
    const float* __restrict__ x,      // input_tensor (B,S,C,H,W)
    const float* __restrict__ pos,    // pos_decoding (B,S,C,H,W)
    const float* __restrict__ wgt,    // conv_w (384,128,3,3,3)
    const float* __restrict__ bias,   // conv_b (384,)
    bf16_t* __restrict__ qws,         // [b][n][s][d][h][w]
    bf16_t* __restrict__ kws,         // [b][n][s][d][h][w]
    bf16_t* __restrict__ vws)         // [b][n][d][s][h][w]
{
  __shared__ bf16_t As[128 * 40];    // A tile [m=co 128][k=ci 32], bf16
  __shared__ bf16_t Bs[32 * 40];     // B tile [n=w 32][k=ci 32], bf16
  __shared__ float  Bf32[32 * 32];   // TDM landing buffer [k=ci 32][w 32], f32

  const int tid  = threadIdx.x;
  const int lane = tid & 31, wid = tid >> 5;
  const int lq = lane & 15, half = lane >> 4;
  const int lo = half * 8;                 // K offset per 16-bit WMMA layout
  const int co0 = blockIdx.x * 128;
  const int od  = blockIdx.y >> 5, oh = blockIdx.y & 31;
  const int b   = blockIdx.z;
  const int mT  = wid >> 1, nT = wid & 1;

  v8f acc0 = vzero8(), acc1 = vzero8();

  for (int c4 = 0; c4 < 4; ++c4) {
    const int c0 = c4 * 32;
    for (int tap = 0; tap < 27; ++tap) {
      const int kd = tap / 9, kh = (tap / 3) % 3, kw = tap % 3;
      const int id = od + kd - 2;
      const int ih = oh + kh - 1;

      // --- stage A tile: weight gathers (stride 27), L2-resident ---
      {
        const int m = tid >> 1;
        const int abase = (co0 + m) * CIK + c0 * 27 + tap;
        #pragma unroll
        for (int j = 0; j < 16; ++j) {
          const int k = (tid & 1) * 16 + j;
          As[m * 40 + k] = (bf16_t)wgt[abase + k * 27];
        }
        if (tap + 1 < 27) __builtin_prefetch(&wgt[abase + 1], 0, 1);
      }

      // --- stage B tile ---
      const bool tdm_ok = (kw == 1) & ((unsigned)id < 32u) &
                          ((unsigned)ih < 32u);     // block-uniform
      if (tdm_ok) {
        // Interior tap: Tensor Data Mover DMA. 2D tile: dim0 = 32 w (4B,
        // contiguous), dim1 = 32 ci lines with stride 1024 elements.
        if (wid == 0) {
          const unsigned long long ga = (unsigned long long)(uintptr_t)
              (x + ((b * NS + id) * NC + c0) * NHW + ih * 32);
          const unsigned lds = (unsigned)(uintptr_t)(&Bf32[0]);
          u32x4 g0 = { 1u,                                   // count=1
                       lds,                                  // lds_addr
                       (unsigned)ga,                         // global_addr lo
                       (unsigned)(ga >> 32) | 0x80000000u }; // addr hi | type=2
          i32x8 g1 = { 0x00020000,   // data_size=4B
                       0x00200000,   // tensor_dim0=32 (low16 @ bit48)
                       0x00200000,   // tensor_dim1=32 (low16 @ bit80)
                       0x00200000,   // tile_dim0=32   (@ bit112)
                       0x00000020,   // tile_dim1=32   (@ bit128)
                       0x00000400,   // tensor_dim0_stride=1024 (@ bit160)
                       0, 0 };
          i32x4 gz4 = {0, 0, 0, 0};
          i32x8 gz8 = {0, 0, 0, 0, 0, 0, 0, 0};
          __builtin_amdgcn_tensor_load_to_lds(g0, g1, gz4, gz4, gz8, 0);
          __builtin_amdgcn_s_wait_tensorcnt(0);
        }
        __syncthreads();                       // Bf32 + As visible
        // transpose-convert f32 [k][w] -> bf16 [w][k]
        #pragma unroll
        for (int j = 0; j < 4; ++j) {
          const int e = tid + 256 * j;
          const int k = e >> 5, w = e & 31;
          Bs[w * 40 + k] = (bf16_t)Bf32[k * 32 + w];
        }
        __syncthreads();                       // Bs ready
      } else {
        // Boundary tap: VALU staging with zero padding, coalesced over w.
        const int wcol = tid & 31, krow = tid >> 5;
        const int iw = wcol + kw - 1;
        const bool inb = ((unsigned)id < 32u) & ((unsigned)ih < 32u) &
                         ((unsigned)iw < 32u);
        #pragma unroll
        for (int j = 0; j < 4; ++j) {
          const int k = krow + 8 * j;
          float xv = 0.f;
          if (inb) xv = x[((b * NS + id) * NC + (c0 + k)) * NHW + ih * 32 + iw];
          Bs[wcol * 40 + k] = (bf16_t)xv;
        }
        __syncthreads();                       // As + Bs ready
      }

      // --- fragments (documented 16-bit WMMA layout) + 2 WMMAs ---
      BF16Frag fa0, fa1, fb;
      const int ar0 = mT * 32 + lq, ar1 = mT * 32 + 16 + lq;
      fa0.h[0] = *(const v8bf*)&As[ar0 * 40 + lo];
      fa0.h[1] = *(const v8bf*)&As[ar0 * 40 + 16 + lo];
      fa1.h[0] = *(const v8bf*)&As[ar1 * 40 + lo];
      fa1.h[1] = *(const v8bf*)&As[ar1 * 40 + 16 + lo];
      const int brow = nT * 16 + lq;
      fb.h[0] = *(const v8bf*)&Bs[brow * 40 + lo];
      fb.h[1] = *(const v8bf*)&Bs[brow * 40 + 16 + lo];
      acc0 = wmma_bf16(fa0.v, fb.v, acc0);
      acc1 = wmma_bf16(fa1.v, fb.v, acc1);
      __syncthreads();
    }
  }

  // Epilogue: + bias (+pos for q,k); scatter bf16 to attention-friendly ws.
  const int wcol = nT * 16 + lq;
  #pragma unroll
  for (int ti = 0; ti < 2; ++ti) {
    const v8f acc = ti ? acc1 : acc0;
    #pragma unroll
    for (int r = 0; r < 8; ++r) {
      const int co = co0 + mT * 32 + ti * 16 + half * 8 + r;
      float val = acc[r] + bias[co];
      const int g = co >> 7;                  // 0=q 1=k 2=v
      const int c = co & 127, n = c >> 5, dc = c & 31;
      if (g < 2) val += pos[((b * NS + od) * NC + c) * NHW + oh * 32 + wcol];
      const bf16_t bv = (bf16_t)val;
      const int bn = b * NHEAD + n;
      if (g == 0)
        qws[((bn * NS + od) * HD + dc) * NHW + oh * 32 + wcol] = bv;
      else if (g == 1)
        kws[((bn * NS + od) * HD + dc) * NHW + oh * 32 + wcol] = bv;
      else
        vws[((bn * HD + dc) * NS + od) * NHW + oh * 32 + wcol] = bv;
    }
  }
}

// ----------------------------------------------------------------------------
// Kernel 2: causal attention per (b, n, h, w). Grid (32, 4, 4), block 256
// (8 waves). Stage 8 w-columns of q/k (later v, reusing q's buffer) into LDS;
// each wave owns one w: scores (4 WMMA), register softmax with 16-lane
// shfl_xor reductions, out = am * v^T (4 WMMA).
// ----------------------------------------------------------------------------
__global__ __launch_bounds__(256) void attn_wmma(
    const bf16_t* __restrict__ qws, const bf16_t* __restrict__ kws,
    const bf16_t* __restrict__ vws, float* __restrict__ out)
{
  __shared__ bf16_t sQ[8][32 * 40];   // q [s][d]; later reused for v [d][s]
  __shared__ bf16_t sK[8][32 * 40];   // k [s][d]
  __shared__ bf16_t sAm[8][32 * 40];  // softmax probs [t][s]
  // total LDS: 61,440 B

  const int tid  = threadIdx.x;
  const int lane = tid & 31, wv = tid >> 5;
  const int lq = lane & 15, half = lane >> 4;
  const int lo = half * 8;
  const int h = blockIdx.x, n = blockIdx.y, b = blockIdx.z;
  const int bn = b * NHEAD + n;
  const float scale = 0.17677669529663687f;   // 1/sqrt(32)

  for (int chunk = 0; chunk < 4; ++chunk) {
    const int w0 = chunk * 8;

    // stage q,k: one uint4 = 8 bf16 w-values per (row,col)
    #pragma unroll
    for (int j = 0; j < 4; ++j) {
      const int p = tid + 256 * j;            // p = s*32+d
      const int g = (bn * 1024 + p) * NHW + h * 32 + w0;
      uint4 qv = *(const uint4*)(qws + g);
      uint4 kv = *(const uint4*)(kws + g);
      const int row = p >> 5, col = p & 31;
      const unsigned short* qp = (const unsigned short*)&qv;
      const unsigned short* kp = (const unsigned short*)&kv;
      #pragma unroll
      for (int wi = 0; wi < 8; ++wi) {
        ((unsigned short*)sQ[wi])[row * 40 + col] = qp[wi];
        ((unsigned short*)sK[wi])[row * 40 + col] = kp[wi];
      }
    }
    __syncthreads();

    // scores = q^T k / sqrt(d): 2x2 tiles, K=d=32
    v8f sc00 = vzero8(), sc01 = vzero8(), sc10 = vzero8(), sc11 = vzero8();
    BF16Frag aT[2], bS[2];
    #pragma unroll
    for (int tI = 0; tI < 2; ++tI) {
      const int row = tI * 16 + lq;
      aT[tI].h[0] = *(const v8bf*)&sQ[wv][row * 40 + lo];
      aT[tI].h[1] = *(const v8bf*)&sQ[wv][row * 40 + 16 + lo];
    }
    #pragma unroll
    for (int sJ = 0; sJ < 2; ++sJ) {
      const int row = sJ * 16 + lq;
      bS[sJ].h[0] = *(const v8bf*)&sK[wv][row * 40 + lo];
      bS[sJ].h[1] = *(const v8bf*)&sK[wv][row * 40 + 16 + lo];
    }
    sc00 = wmma_bf16(aT[0].v, bS[0].v, sc00);
    sc01 = wmma_bf16(aT[0].v, bS[1].v, sc01);
    sc10 = wmma_bf16(aT[1].v, bS[0].v, sc10);
    sc11 = wmma_bf16(aT[1].v, bS[1].v, sc11);

    // causal mask + (am==0 -> -1e9) + softmax over s, in registers.
    #pragma unroll
    for (int tI = 0; tI < 2; ++tI) {
      #pragma unroll
      for (int r = 0; r < 8; ++r) {
        const int t = tI * 16 + half * 8 + r;
        float a0 = (tI ? sc10[r] : sc00[r]) * scale;   // s = lq
        float a1 = (tI ? sc11[r] : sc01[r]) * scale;   // s = 16+lq
        const int s0 = lq, s1 = 16 + lq;
        a0 = (s0 <= t) ? a0 : 0.f; if (a0 == 0.f) a0 = -1.0e9f;
        a1 = (s1 <= t) ? a1 : 0.f; if (a1 == 0.f) a1 = -1.0e9f;
        float mx = fmaxf(a0, a1);
        #pragma unroll
        for (int d = 1; d < 16; d <<= 1) mx = fmaxf(mx, __shfl_xor(mx, d, 32));
        const float e0 = __expf(a0 - mx), e1 = __expf(a1 - mx);
        float sum = e0 + e1;
        #pragma unroll
        for (int d = 1; d < 16; d <<= 1) sum += __shfl_xor(sum, d, 32);
        const float inv = 1.f / sum;
        sAm[wv][t * 40 + s0] = (bf16_t)(e0 * inv);
        sAm[wv][t * 40 + s1] = (bf16_t)(e1 * inv);
      }
    }
    __syncthreads();

    // restage v into sQ: p = d*32+s -> [d][s]
    #pragma unroll
    for (int j = 0; j < 4; ++j) {
      const int p = tid + 256 * j;
      const int g = (bn * 1024 + p) * NHW + h * 32 + w0;
      uint4 vvv = *(const uint4*)(vws + g);
      const int row = p >> 5, col = p & 31;
      const unsigned short* vp = (const unsigned short*)&vvv;
      #pragma unroll
      for (int wi = 0; wi < 8; ++wi)
        ((unsigned short*)sQ[wi])[row * 40 + col] = vp[wi];
    }
    __syncthreads();

    // out[t][dd] = sum_s am[t][s] * v[dd][s]: 2x2 tiles, K=s=32
    #pragma unroll
    for (int tI = 0; tI < 2; ++tI) {
      BF16Frag fa;
      const int row = tI * 16 + lq;
      fa.h[0] = *(const v8bf*)&sAm[wv][row * 40 + lo];
      fa.h[1] = *(const v8bf*)&sAm[wv][row * 40 + 16 + lo];
      #pragma unroll
      for (int dJ = 0; dJ < 2; ++dJ) {
        BF16Frag fb;
        const int vrow = dJ * 16 + lq;
        fb.h[0] = *(const v8bf*)&sQ[wv][vrow * 40 + lo];
        fb.h[1] = *(const v8bf*)&sQ[wv][vrow * 40 + 16 + lo];
        v8f oc = vzero8();
        oc = wmma_bf16(fa.v, fb.v, oc);
        const int dd = dJ * 16 + lq;
        #pragma unroll
        for (int r = 0; r < 8; ++r) {
          const int t = tI * 16 + half * 8 + r;
          out[((b * NS + t) * NC + (n * HD + dd)) * NHW + h * 32 + w0 + wv] =
              oc[r];
        }
      }
    }
    __syncthreads();
  }
}

// ----------------------------------------------------------------------------
// Launch. Workspace: 96 MiB bf16 qkv (q|k|v, 32 MiB each), L2-resident.
// ----------------------------------------------------------------------------
extern "C" void kernel_launch(void* const* d_in, const int* in_sizes, int n_in,
                              void* d_out, int out_size, void* d_ws,
                              size_t ws_size, hipStream_t stream) {
  const float* x    = (const float*)d_in[0];
  const float* pos  = (const float*)d_in[1];
  const float* wgt  = (const float*)d_in[2];
  const float* bias = (const float*)d_in[3];
  float* out = (float*)d_out;

  bf16_t* qws = (bf16_t*)d_ws;
  bf16_t* kws = qws + WS_TENSOR_ELEMS;
  bf16_t* vws = kws + WS_TENSOR_ELEMS;

  conv_qkv_wmma<<<dim3(NCO / 128, NS * NH, NB), 256, 0, stream>>>(
      x, pos, wgt, bias, qws, kws, vws);
  attn_wmma<<<dim3(NH, NHEAD, NB), 256, 0, stream>>>(qws, kws, vws, out);
}
